// UpsampleBlock_420906795569
// MI455X (gfx1250) — matrix operand order repaired and checked by reading, not compiled, not run on hardware
//
#include <hip/hip_runtime.h>
#include <hip/hip_bf16.h>

typedef __attribute__((ext_vector_type(16))) __bf16 v16bf;
typedef __attribute__((ext_vector_type(8)))  __bf16 v8bf;
typedef __attribute__((ext_vector_type(8)))  float  v8f;

#define CIN  256
#define COUT 256

// ---------------------------------------------------------------------------
// LayerNorm -> bf16.  One wave32 per row of 256 floats; 8 elems per lane.
// ---------------------------------------------------------------------------
__global__ void ln_bf16_kernel(const float* __restrict__ X,
                               const float* __restrict__ g,
                               const float* __restrict__ b,
                               __bf16* __restrict__ Y, int nrows) {
    const int lane = threadIdx.x & 31;
    const int wave = threadIdx.x >> 5;
    const int row  = blockIdx.x * 8 + wave;
    if (row >= nrows) return;

    const float* xr = X + (size_t)row * CIN;
    const int c0 = lane * 8;
    float4 v0 = *(const float4*)(xr + c0);
    float4 v1 = *(const float4*)(xr + c0 + 4);

    float s  = v0.x + v0.y + v0.z + v0.w + v1.x + v1.y + v1.z + v1.w;
    float ss = fmaf(v0.x, v0.x, fmaf(v0.y, v0.y, fmaf(v0.z, v0.z, fmaf(v0.w, v0.w,
               fmaf(v1.x, v1.x, fmaf(v1.y, v1.y, fmaf(v1.z, v1.z, v1.w * v1.w)))))));
    #pragma unroll
    for (int m = 16; m >= 1; m >>= 1) {
        s  += __shfl_xor(s,  m, 32);
        ss += __shfl_xor(ss, m, 32);
    }
    const float mu   = s * (1.0f / CIN);
    const float var  = ss * (1.0f / CIN) - mu * mu;
    const float rstd = rsqrtf(var + 1e-5f);

    float4 g0 = *(const float4*)(g + c0);
    float4 g1 = *(const float4*)(g + c0 + 4);
    float4 b0 = *(const float4*)(b + c0);
    float4 b1 = *(const float4*)(b + c0 + 4);

    v8bf out;
    out[0] = (__bf16)(fmaf((v0.x - mu) * rstd, g0.x, b0.x));
    out[1] = (__bf16)(fmaf((v0.y - mu) * rstd, g0.y, b0.y));
    out[2] = (__bf16)(fmaf((v0.z - mu) * rstd, g0.z, b0.z));
    out[3] = (__bf16)(fmaf((v0.w - mu) * rstd, g0.w, b0.w));
    out[4] = (__bf16)(fmaf((v1.x - mu) * rstd, g1.x, b1.x));
    out[5] = (__bf16)(fmaf((v1.y - mu) * rstd, g1.y, b1.y));
    out[6] = (__bf16)(fmaf((v1.z - mu) * rstd, g1.z, b1.z));
    out[7] = (__bf16)(fmaf((v1.w - mu) * rstd, g1.w, b1.w));
    *(v8bf*)(Y + (size_t)row * CIN + c0) = out;
}

// ---------------------------------------------------------------------------
// W [K,N] f32 -> Wt [N,K] bf16  (so WMMA B operand loads are contiguous in K)
// ---------------------------------------------------------------------------
__global__ void transpose_bf16_kernel(const float* __restrict__ W,
                                      __bf16* __restrict__ Wt) {
    const int t = blockIdx.x * blockDim.x + threadIdx.x; // 0..65535
    const int n = t >> 8;
    const int k = t & 255;
    Wt[(size_t)n * CIN + k] = (__bf16)W[(size_t)k * COUT + n];
}

// ---------------------------------------------------------------------------
// C[M,256] = A_bf16[M,256] @ Wt_bf16[256,256]^T + bias.
// One 16x16 tile per wave; 8 waves per block; K fully unrolled (8 WMMAs).
// A fragment (16-bit A 16x32): lane<16 holds K{0..15},
// lane>=16 holds K{16..31}, M = lane&15.
// B fragment: lane<16 holds K{0..15}, lane>=16 K{16..31}, N = lane&15.
// ---------------------------------------------------------------------------
__global__ void gemm_bias_kernel(const __bf16* __restrict__ A,
                                 const __bf16* __restrict__ Bt,
                                 const float* __restrict__ bias,
                                 float* __restrict__ C, int M) {
    const int lane = threadIdx.x & 31;
    const int wave = threadIdx.x >> 5;
    const int tile = blockIdx.x * 8 + wave;
    const int nt   = tile & 15;          // N/16 == 16 tiles
    const int mt   = tile >> 4;
    if (mt * 16 >= M) return;

    const int l15 = lane & 15;
    const int hi  = lane >> 4;

    const __bf16* arow = A  + (size_t)(mt * 16 + l15) * CIN;
    const __bf16* brow = Bt + (size_t)(nt * 16 + l15) * CIN;

    v8f acc = {};
    #pragma unroll
    for (int k = 0; k < CIN; k += 32) {
        v8bf alo = *(const v8bf*)(arow + k + 16 * hi);
        v8bf ahi = *(const v8bf*)(arow + k + 16 * hi + 8);
        v16bf a  = __builtin_shufflevector(alo, ahi,
                       0,1,2,3,4,5,6,7,8,9,10,11,12,13,14,15);
        v16bf bf = *(const v16bf*)(brow + k + 16 * hi);
        acc = __builtin_amdgcn_wmma_f32_16x16x32_bf16(
                  false, a, false, bf, (short)0, acc, false, false);
    }

    const float bv = bias[nt * 16 + l15];
    float* crow = C + (size_t)(mt * 16 + 8 * hi) * COUT + nt * 16 + l15;
    #pragma unroll
    for (int i = 0; i < 8; ++i)
        crow[(size_t)i * COUT] = acc[i] + bv;
}

// ---------------------------------------------------------------------------
// Top-3 nearest coarse points per fine point. Coarse positions staged in LDS
// (48 KB of the 320 KB WGP pool). One fine point per thread.
// ---------------------------------------------------------------------------
__global__ void knn3_kernel(const float* __restrict__ up_pos,
                            const float* __restrict__ c_pos,
                            int* __restrict__ kidx,
                            float* __restrict__ kw) {
    __shared__ float sx[4096];
    __shared__ float sy[4096];
    __shared__ float sz[4096];
    for (int i = threadIdx.x; i < 4096; i += 256) {
        sx[i] = c_pos[3 * i + 0];
        sy[i] = c_pos[3 * i + 1];
        sz[i] = c_pos[3 * i + 2];
    }
    __syncthreads();

    const int u = blockIdx.x * 256 + threadIdx.x;
    const float px = up_pos[3 * u + 0];
    const float py = up_pos[3 * u + 1];
    const float pz = up_pos[3 * u + 2];

    float d0 = 3.4e38f, d1 = 3.4e38f, d2 = 3.4e38f;
    int   i0 = 0,       i1 = 0,       i2 = 0;
    for (int c = 0; c < 4096; ++c) {
        const float dx = px - sx[c];
        const float dy = py - sy[c];
        const float dz = pz - sz[c];
        const float d  = fmaf(dx, dx, fmaf(dy, dy, dz * dz));
        if (d < d2) {
            if (d < d0)      { d2 = d1; i2 = i1; d1 = d0; i1 = i0; d0 = d; i0 = c; }
            else if (d < d1) { d2 = d1; i2 = i1; d1 = d;  i1 = c; }
            else             { d2 = d;  i2 = c; }
        }
    }
    float w0 = 1.0f / (d0 + 1e-8f);
    float w1 = 1.0f / (d1 + 1e-8f);
    float w2 = 1.0f / (d2 + 1e-8f);
    const float inv = 1.0f / (w0 + w1 + w2);
    kidx[u * 3 + 0] = i0;  kw[u * 3 + 0] = w0 * inv;
    kidx[u * 3 + 1] = i1;  kw[u * 3 + 1] = w1 * inv;
    kidx[u * 3 + 2] = i2;  kw[u * 3 + 2] = w2 * inv;
}

// ---------------------------------------------------------------------------
// out[u,:] += w0*x[i0,:] + w1*x[i1,:] + w2*x[i2,:]   (out already holds xu)
// 4 rows per 256-thread block; float4 per thread. x (4 MB) lives in L2.
// ---------------------------------------------------------------------------
__global__ void interp_add_kernel(float* __restrict__ out,
                                  const float* __restrict__ x,
                                  const int* __restrict__ kidx,
                                  const float* __restrict__ kw) {
    const int t = threadIdx.x;
    const int u = blockIdx.x * 4 + (t >> 6);
    const int c = (t & 63) * 4;

    const int   i0 = kidx[u * 3 + 0], i1 = kidx[u * 3 + 1], i2 = kidx[u * 3 + 2];
    const float w0 = kw[u * 3 + 0],   w1 = kw[u * 3 + 1],   w2 = kw[u * 3 + 2];

    float4 a = *(const float4*)(x + (size_t)i0 * COUT + c);
    float4 b = *(const float4*)(x + (size_t)i1 * COUT + c);
    float4 d = *(const float4*)(x + (size_t)i2 * COUT + c);
    float4 o = *(float4*)(out + (size_t)u * COUT + c);
    o.x = fmaf(w0, a.x, fmaf(w1, b.x, fmaf(w2, d.x, o.x)));
    o.y = fmaf(w0, a.y, fmaf(w1, b.y, fmaf(w2, d.y, o.y)));
    o.z = fmaf(w0, a.z, fmaf(w1, b.z, fmaf(w2, d.z, o.z)));
    o.w = fmaf(w0, a.w, fmaf(w1, b.w, fmaf(w2, d.w, o.w)));
    *(float4*)(out + (size_t)u * COUT + c) = o;
}

// ---------------------------------------------------------------------------
extern "C" void kernel_launch(void* const* d_in, const int* in_sizes, int n_in,
                              void* d_out, int out_size, void* d_ws, size_t ws_size,
                              hipStream_t stream) {
    const float* x_feat   = (const float*)d_in[0];   // [4096,256]
    const float* x_pos    = (const float*)d_in[1];   // [4096,3]
    const float* xup_feat = (const float*)d_in[2];   // [16384,256]
    const float* xup_pos  = (const float*)d_in[3];   // [16384,3]
    const float* g1       = (const float*)d_in[4];
    const float* b1       = (const float*)d_in[5];
    const float* W        = (const float*)d_in[6];   // [256,256]
    const float* bb       = (const float*)d_in[7];
    const float* g2       = (const float*)d_in[8];
    const float* b2       = (const float*)d_in[9];
    const float* W_up     = (const float*)d_in[10];  // [256,256]
    const float* b_up     = (const float*)d_in[11];

    const int Nc = 4096, Nu = 16384;
    float* out = (float*)d_out;                      // [16384,256]

    // workspace carve-up (all offsets 256B-aligned)
    char* ws = (char*)d_ws;
    size_t off = 0;
    __bf16* xln  = (__bf16*)(ws + off); off += (size_t)Nc * CIN * 2;   // 2 MB
    __bf16* xuln = (__bf16*)(ws + off); off += (size_t)Nu * CIN * 2;   // 8 MB
    __bf16* Wt   = (__bf16*)(ws + off); off += (size_t)CIN * COUT * 2; // 128 KB
    __bf16* Wut  = (__bf16*)(ws + off); off += (size_t)CIN * COUT * 2; // 128 KB
    float*  xc   = (float*) (ws + off); off += (size_t)Nc * COUT * 4;  // 4 MB
    int*    kidx = (int*)   (ws + off); off += (size_t)Nu * 3 * 4;     // 192 KB
    float*  kw   = (float*) (ws + off); off += (size_t)Nu * 3 * 4;     // 192 KB

    // 1) LayerNorms -> bf16
    ln_bf16_kernel<<<Nc / 8, 256, 0, stream>>>(x_feat, g1, b1, xln, Nc);
    ln_bf16_kernel<<<Nu / 8, 256, 0, stream>>>(xup_feat, g2, b2, xuln, Nu);

    // 2) weight transposes -> bf16
    transpose_bf16_kernel<<<(CIN * COUT) / 256, 256, 0, stream>>>(W, Wt);
    transpose_bf16_kernel<<<(CIN * COUT) / 256, 256, 0, stream>>>(W_up, Wut);

    // 3) GEMMs via v_wmma_f32_16x16x32_bf16
    gemm_bias_kernel<<<(Nc / 16) * 16 / 8, 256, 0, stream>>>(xln, Wt, bb, xc, Nc);
    gemm_bias_kernel<<<(Nu / 16) * 16 / 8, 256, 0, stream>>>(xuln, Wut, b_up, out, Nu);

    // 4) top-3 nearest neighbors + weights
    knn3_kernel<<<Nu / 256, 256, 0, stream>>>(xup_pos, x_pos, kidx, kw);

    // 5) out += weighted gather of coarse features
    interp_add_kernel<<<Nu / 4, 256, 0, stream>>>(out, xc, kidx, kw);
}
